// VisionTransformer_84413287235956
// MI455X (gfx1250) — matrix operand order repaired and checked by reading, not compiled, not run on hardware
//
#include <hip/hip_runtime.h>
#include <hip/hip_bf16.h>
#include <hip/hip_fp16.h>

typedef _Float16 h16;
typedef __attribute__((ext_vector_type(16))) _Float16 v16h;
typedef __attribute__((ext_vector_type(8)))  float    v8f;
typedef __attribute__((ext_vector_type(4)))  unsigned int v4u;
typedef __attribute__((ext_vector_type(8)))  int      v8i;
typedef __attribute__((ext_vector_type(4)))  int      v4i;
typedef unsigned int u32;

#define DMODEL 384
#define NHEAD 6
#define HDIM 64
#define NLAYER 4
#define BATCH 8
#define NPATCH 1024
#define SEQLEN 1025
#define NPAD 1056               /* 66*16, 33*32 */
#define MROWS (BATCH*NPAD)      /* 8448 = 66*128 */
#define DFF 1536
#define NCLS 2
#define PHH 16
#define PWW 16

#if defined(__gfx1250__) && __has_builtin(__builtin_amdgcn_tensor_load_to_lds)
#define HAVE_TDM 1
#if __has_include(<hip/amd_detail/amd_gfx1250_TDM.h>)
#define TDM_6ARG 1
#else
#define TDM_6ARG 0
#endif
#else
#define HAVE_TDM 0
#endif

__device__ __forceinline__ v8f zero8() {
  v8f z;
#pragma unroll
  for (int r = 0; r < 8; ++r) z[r] = 0.0f;
  return z;
}

/* ----------------------- weight packing / conversion ----------------------- */

__global__ void cvt_f16_kernel(const float* __restrict__ src, h16* __restrict__ dst, int n) {
  int i = blockIdx.x * blockDim.x + threadIdx.x;
  if (i < n) dst[i] = (h16)src[i];
}

/* Wqkv[l][d][j], j = sel*384 + h*64 + e  (sel: 0=q,1=k,2=v); Bqkv[l][j] */
__global__ void pack_qkv_kernel(const float* __restrict__ wq, const float* __restrict__ wk,
                                const float* __restrict__ wv, const float* __restrict__ bq,
                                const float* __restrict__ bk, const float* __restrict__ bv,
                                h16* __restrict__ Wqkv, float* __restrict__ Bqkv) {
  long long idx = (long long)blockIdx.x * blockDim.x + threadIdx.x;
  const long long total = (long long)NLAYER * DMODEL * 3 * DMODEL;
  if (idx >= total) return;
  int l = (int)(idx / (DMODEL * 3 * DMODEL));
  int rem = (int)(idx % (DMODEL * 3 * DMODEL));
  int d = rem / (3 * DMODEL);
  int j = rem % (3 * DMODEL);
  int sel = j / DMODEL;
  int jj = j % DMODEL;
  int h = jj / HDIM, e = jj % HDIM;
  const float* w = (sel == 0) ? wq : ((sel == 1) ? wk : wv);
  float val = w[(((long long)l * NHEAD + h) * DMODEL + d) * HDIM + e];
  Wqkv[idx] = (h16)val;
  if (d == 0) {
    const float* bb = (sel == 0) ? bq : ((sel == 1) ? bk : bv);
    Bqkv[l * 3 * DMODEL + j] = bb[((long long)l * NHEAD + h) * HDIM + e];
  }
}

/* ------------------------- patch embed + pos enc --------------------------- */

__global__ __launch_bounds__(DMODEL)
void embed_kernel(const float* __restrict__ images, const float* __restrict__ patch_w,
                  const float* __restrict__ patch_b, const float* __restrict__ cls_tok,
                  float* __restrict__ X) {
  int row = blockIdx.x;
  int b = row / NPAD, t = row % NPAD;
  int col = threadIdx.x;
  __shared__ float img[256];
  float acc;
  if (t >= 1 && t <= NPATCH) {
    if (col < 256) img[col] = images[((long long)b * NPATCH + (t - 1)) * 256 + col];
    __syncthreads();
    float s = patch_b[col];
#pragma unroll 8
    for (int k = 0; k < 256; ++k) s += img[k] * patch_w[k * DMODEL + col];
    acc = s;
  } else if (t == 0) {
    acc = cls_tok[col];
  } else {
    X[(long long)row * DMODEL + col] = 0.0f;
    return;
  }
  float fi = (float)(col & ~1);
  float ang = (float)t * __expf(-(fi / (float)DMODEL) * 9.210340371976184f);
  float pe = (col & 1) ? __cosf(ang) : __sinf(ang);
  X[(long long)row * DMODEL + col] = acc + pe;
}

/* -------------------------------- layernorm -------------------------------- */

__global__ __launch_bounds__(128)
void ln_kernel(const float* __restrict__ X, const float* __restrict__ g,
               const float* __restrict__ bvec, h16* __restrict__ Hout) {
  int row = blockIdx.x;
  int tid = threadIdx.x;
  int wave = tid >> 5, lane = tid & 31;
  __shared__ float red[8];
  const float* xr = X + (long long)row * DMODEL;
  float v0 = xr[tid], v1 = xr[tid + 128], v2 = xr[tid + 256];
  float s = v0 + v1 + v2;
#pragma unroll
  for (int o = 16; o > 0; o >>= 1) s += __shfl_xor(s, o, 32);
  if (lane == 0) red[wave] = s;
  __syncthreads();
  float mu = (red[0] + red[1] + red[2] + red[3]) * (1.0f / DMODEL);
  float d0 = v0 - mu, d1 = v1 - mu, d2 = v2 - mu;
  float q = d0 * d0 + d1 * d1 + d2 * d2;
#pragma unroll
  for (int o = 16; o > 0; o >>= 1) q += __shfl_xor(q, o, 32);
  if (lane == 0) red[4 + wave] = q;
  __syncthreads();
  float var = (red[4] + red[5] + red[6] + red[7]) * (1.0f / DMODEL);
  float r = rsqrtf(var + 1e-5f);
  h16* ho = Hout + (long long)row * DMODEL;
  ho[tid]       = (h16)(d0 * r * g[tid]       + bvec[tid]);
  ho[tid + 128] = (h16)(d1 * r * g[tid + 128] + bvec[tid + 128]);
  ho[tid + 256] = (h16)(d2 * r * g[tid + 256] + bvec[tid + 256]);
}

/* ------------------------------ WMMA GEMM ---------------------------------- */
/* C = A(f16, MxK row-major) * B(f16, KxN row-major) + bias ; fused epilogues  */

#define EP_F16 0
#define EP_F16_GELU 1
#define EP_F32_RESID 2

#define BM 128
#define BN 64
#define AST 40   /* A LDS stride: 32 + 4-dword pad -> 16B aligned rows */
#define BTST 34  /* transposed-B LDS stride */

#if HAVE_TDM
/* ---- TDM descriptor-based async A-tile loader (CDNA5 ISA) -- */

__device__ __forceinline__ void tdm_issue(u32 lds_off, const h16* g,
                                          u32 tile_w, u32 tile_h,
                                          u32 stride_elems, u32 flags) {
  union { u32 w[4]; v4u v; } g0;
  union { u32 w[8]; v8i v; } g1;
  unsigned long long ga = (unsigned long long)(size_t)g;
  g0.w[0] = 1u;                                   /* count=1, user mode */
  g0.w[1] = lds_off;                              /* lds_addr */
  g0.w[2] = (u32)ga;                              /* global_addr[31:0] */
  g0.w[3] = (u32)(ga >> 32) | 0x80000000u;        /* global_addr[56:32] | type=2 */
  g1.w[0] = flags;                                /* mask=0|data_size|pad cfg */
  g1.w[1] = (stride_elems & 0xFFFFu) << 16;       /* tensor_dim0 lo16 */
  g1.w[2] = (stride_elems >> 16) & 0xFFFFu;       /* tensor_dim0 hi16; dim1 lo16=0 */
  g1.w[3] = 0x0010u | (tile_w << 16);             /* dim1 hi16 (1M rows); tile_dim0 */
  g1.w[4] = tile_h & 0xFFFFu;                     /* tile_dim1; tile_dim2=0 */
  g1.w[5] = stride_elems;                         /* tensor_dim0_stride lo32 */
  g1.w[6] = 0;
  g1.w[7] = 0;
  v4i z4 = {0, 0, 0, 0};
#if TDM_6ARG
  v8i z8 = {0, 0, 0, 0, 0, 0, 0, 0};
  __builtin_amdgcn_tensor_load_to_lds(g0.v, g1.v, z4, z4, z8, 0);
#else
  __builtin_amdgcn_tensor_load_to_lds(g0.v, g1.v, z4, z4, 0);
#endif
}

__global__ __launch_bounds__(256)
void gemm_kernel(const h16* __restrict__ A, const h16* __restrict__ Bw,
                 const float* __restrict__ bias, const float* __restrict__ resid,
                 float* __restrict__ outF32, h16* __restrict__ outF16,
                 int Nc, int Kd, int mode) {
  __shared__ h16 As[2][BM * AST];
  __shared__ h16 BsT[2][BN * BTST];
  int tid = threadIdx.x;
  int bm = blockIdx.y * BM;
  int bn = blockIdx.x * BN;
  int wave = tid >> 5, lane = tid & 31;
  int wm = wave & 3, wn = wave >> 2;

  v8f acc[2][2];
#pragma unroll
  for (int i = 0; i < 2; ++i)
#pragma unroll
    for (int j = 0; j < 2; ++j) acc[i][j] = zero8();

  int mfrag = lane & 15;
  int kbase2 = (lane < 16) ? 0 : 4;
  int khalf = (lane < 16) ? 0 : 16;

  /* pad cfg: interval code c -> 2^(c+1) DWORDs; amount code a -> a+1 DWORDs */
  const u32 FA = (1u << 16) | (1u << 20) | (3u << 22) | (3u << 25); /* 16 dw + 4 dw pad */

  /* stage transposed B tile for K-step k0 into buffer `buf` (weights, L2-hot) */
  auto stageB = [&](int buf, int k0) {
    int r = tid >> 3;
    int c0 = (tid & 7) * 8;
    union { uint4 q; h16 h[8]; } pk;
    pk.q = *(const uint4*)(Bw + (long long)(k0 + r) * Nc + bn + c0);
#pragma unroll
    for (int j = 0; j < 8; ++j) BsT[buf][(c0 + j) * BTST + r] = pk.h[j];
  };

  if (wave == 0)
    tdm_issue((u32)(size_t)&As[0][0], A + (size_t)bm * Kd, 32u, (u32)BM, (u32)Kd, FA);
  stageB(0, 0);

  int cur = 0;
  for (int k0 = 0; k0 < Kd; k0 += 32) {
    if (wave == 0) __builtin_amdgcn_s_wait_tensorcnt(0);
    __syncthreads();
    if (k0 + 32 < Kd) {
      if (wave == 0)
        tdm_issue((u32)(size_t)&As[cur ^ 1][0], A + (size_t)bm * Kd + (k0 + 32),
                  32u, (u32)BM, (u32)Kd, FA);
      stageB(cur ^ 1, k0 + 32);
    }
    v16h afrag[2], bfrag[2];
#pragma unroll
    for (int i = 0; i < 2; ++i) {
      const u32* base = (const u32*)&As[cur][(wm * 32 + i * 16 + mfrag) * AST];
      u32* au = (u32*)&afrag[i];
#pragma unroll
      for (int v = 0; v < 4; ++v) {
        au[v]     = base[2 * kbase2 + v];
        au[v + 4] = base[2 * kbase2 + 4 + v];
      }
    }
#pragma unroll
    for (int j = 0; j < 2; ++j) {
      const u32* bb = (const u32*)&BsT[cur][(wn * 32 + j * 16 + mfrag) * BTST];
      u32* bu = (u32*)&bfrag[j];
#pragma unroll
      for (int v = 0; v < 8; ++v) bu[v] = bb[(khalf >> 1) + v];
    }
#pragma unroll
    for (int i = 0; i < 2; ++i)
#pragma unroll
      for (int j = 0; j < 2; ++j)
        acc[i][j] = __builtin_amdgcn_wmma_f32_16x16x32_f16(
            false, afrag[i], false, bfrag[j], (short)0, acc[i][j], false, false);
    cur ^= 1;
  }

  int rowhalf = (lane < 16) ? 0 : 8;
  int ncol = lane & 15;
#pragma unroll
  for (int i = 0; i < 2; ++i)
#pragma unroll
    for (int j = 0; j < 2; ++j) {
      int col = bn + wn * 32 + j * 16 + ncol;
      float bvv = bias ? bias[col] : 0.0f;
#pragma unroll
      for (int r = 0; r < 8; ++r) {
        long long row = bm + wm * 32 + i * 16 + rowhalf + r;
        float v = acc[i][j][r] + bvv;
        if (mode == EP_F32_RESID) {
          outF32[row * Nc + col] = resid[row * Nc + col] + v;
        } else if (mode == EP_F16) {
          outF16[row * Nc + col] = (h16)v;
        } else {
          float gv = 0.5f * v * (1.0f + erff(v * 0.70710678118f));
          outF16[row * Nc + col] = (h16)gv;
        }
      }
    }
}

#else  /* !HAVE_TDM: synchronous staging fallback */

__global__ __launch_bounds__(256)
void gemm_kernel(const h16* __restrict__ A, const h16* __restrict__ Bw,
                 const float* __restrict__ bias, const float* __restrict__ resid,
                 float* __restrict__ outF32, h16* __restrict__ outF16,
                 int Nc, int Kd, int mode) {
  __shared__ h16 As[BM * AST];
  __shared__ h16 BsT[BN * BTST];
  int tid = threadIdx.x;
  int bm = blockIdx.y * BM;
  int bn = blockIdx.x * BN;
  int wave = tid >> 5, lane = tid & 31;
  int wm = wave & 3, wn = wave >> 2;

  v8f acc[2][2];
#pragma unroll
  for (int i = 0; i < 2; ++i)
#pragma unroll
    for (int j = 0; j < 2; ++j) acc[i][j] = zero8();

  int mfrag = lane & 15;
  int kbase = (lane < 16) ? 0 : 8;
  int khalf = (lane < 16) ? 0 : 16;

  for (int k0 = 0; k0 < Kd; k0 += 32) {
#pragma unroll
    for (int c = 0; c < 2; ++c) {
      int chunk = tid * 2 + c;
      int r = chunk >> 2;
      int kk = (chunk & 3) * 8;
      *(uint4*)(&As[r * AST + kk]) =
          *(const uint4*)(A + (long long)(bm + r) * Kd + k0 + kk);
    }
    {
      int r = tid >> 3;
      int c0 = (tid & 7) * 8;
      union { uint4 q; h16 h[8]; } pk;
      pk.q = *(const uint4*)(Bw + (long long)(k0 + r) * Nc + bn + c0);
#pragma unroll
      for (int j = 0; j < 8; ++j) BsT[(c0 + j) * BTST + r] = pk.h[j];
    }
    __syncthreads();

    v16h afrag[2], bfrag[2];
#pragma unroll
    for (int i = 0; i < 2; ++i) {
      const u32* base = (const u32*)&As[(wm * 32 + i * 16 + mfrag) * AST];
      u32* au = (u32*)&afrag[i];
#pragma unroll
      for (int v = 0; v < 4; ++v) {
        au[v]     = base[kbase + v];
        au[v + 4] = base[kbase + 4 + v];
      }
    }
#pragma unroll
    for (int j = 0; j < 2; ++j) {
      const u32* bb = (const u32*)&BsT[(wn * 32 + j * 16 + mfrag) * BTST];
      u32* bu = (u32*)&bfrag[j];
#pragma unroll
      for (int v = 0; v < 8; ++v) bu[v] = bb[(khalf >> 1) + v];
    }
#pragma unroll
    for (int i = 0; i < 2; ++i)
#pragma unroll
      for (int j = 0; j < 2; ++j)
        acc[i][j] = __builtin_amdgcn_wmma_f32_16x16x32_f16(
            false, afrag[i], false, bfrag[j], (short)0, acc[i][j], false, false);
    __syncthreads();
  }

  int rowhalf = (lane < 16) ? 0 : 8;
  int ncol = lane & 15;
#pragma unroll
  for (int i = 0; i < 2; ++i)
#pragma unroll
    for (int j = 0; j < 2; ++j) {
      int col = bn + wn * 32 + j * 16 + ncol;
      float bvv = bias ? bias[col] : 0.0f;
#pragma unroll
      for (int r = 0; r < 8; ++r) {
        long long row = bm + wm * 32 + i * 16 + rowhalf + r;
        float v = acc[i][j][r] + bvv;
        if (mode == EP_F32_RESID) {
          outF32[row * Nc + col] = resid[row * Nc + col] + v;
        } else if (mode == EP_F16) {
          outF16[row * Nc + col] = (h16)v;
        } else {
          float gv = 0.5f * v * (1.0f + erff(v * 0.70710678118f));
          outF16[row * Nc + col] = (h16)gv;
        }
      }
    }
}
#endif /* HAVE_TDM */

/* --------------------------- flash attention ------------------------------- */

__global__ __launch_bounds__(128)
void attn_kernel(const h16* __restrict__ QKV, h16* __restrict__ O) {
  __shared__ h16 P[4][16 * 32];
  int wave = threadIdx.x >> 5, lane = threadIdx.x & 31;
  int blk = blockIdx.x;
  int qt = (blk % 17) * 4 + wave;
  int bh = blk / 17;
  int h = bh % NHEAD, b = bh / NHEAD;
  if (qt >= NPAD / 16) return;

  int mq = lane & 15;
  int kb = (lane < 16) ? 0 : 8;
  int khalf = (lane < 16) ? 0 : 16;
  int nk = lane & 15;
  long long rowbase = (long long)b * NPAD + qt * 16;

  v16h aq[2];
  {
    const h16* qrow = QKV + (rowbase + mq) * (3 * DMODEL) + h * HDIM;
    u32* a0 = (u32*)&aq[0];
    u32* a1 = (u32*)&aq[1];
#pragma unroll
    for (int v = 0; v < 4; ++v) {
      a0[v]     = *(const u32*)(qrow + 2 * kb + 2 * v);
      a0[v + 4] = *(const u32*)(qrow + 2 * kb + 8 + 2 * v);
      a1[v]     = *(const u32*)(qrow + 32 + 2 * kb + 2 * v);
      a1[v + 4] = *(const u32*)(qrow + 32 + 2 * kb + 8 + 2 * v);
    }
  }

  v8f oacc[4];
#pragma unroll
  for (int t = 0; t < 4; ++t) oacc[t] = zero8();
  float mrun[8], srun[8];
#pragma unroll
  for (int r = 0; r < 8; ++r) { mrun[r] = -1e30f; srun[r] = 0.0f; }
  h16* pbuf = P[wave];
  const h16* base_bh = QKV + (long long)b * NPAD * (3 * DMODEL);

  for (int kt0 = 0; kt0 < NPAD; kt0 += 32) {
    v8f s01[2];
#pragma unroll
    for (int half = 0; half < 2; ++half) {
      int tok = kt0 + half * 16 + nk;
      const h16* krow = base_bh + (long long)tok * (3 * DMODEL) + DMODEL + h * HDIM;
      v16h bk0, bk1;
      u32* p0 = (u32*)&bk0;
      u32* p1 = (u32*)&bk1;
#pragma unroll
      for (int v = 0; v < 8; ++v) {
        p0[v] = *(const u32*)(krow + khalf + 2 * v);
        p1[v] = *(const u32*)(krow + 32 + khalf + 2 * v);
      }
      v8f s = zero8();
      s = __builtin_amdgcn_wmma_f32_16x16x32_f16(false, aq[0], false, bk0, (short)0, s, false, false);
      s = __builtin_amdgcn_wmma_f32_16x16x32_f16(false, aq[1], false, bk1, (short)0, s, false, false);
      s01[half] = s;
    }

    int tok0 = kt0 + nk, tok1 = kt0 + 16 + nk;
#pragma unroll
    for (int r = 0; r < 8; ++r) {
      float a = s01[0][r] * 0.125f;
      float c = s01[1][r] * 0.125f;
      if (tok0 >= SEQLEN) a = -1e30f;
      if (tok1 >= SEQLEN) c = -1e30f;
      float mt = fmaxf(a, c);
#pragma unroll
      for (int o = 8; o > 0; o >>= 1) mt = fmaxf(mt, __shfl_xor(mt, o, 32));
      float mnew = fmaxf(mrun[r], mt);
      float alpha = __expf(mrun[r] - mnew);
      float pa = __expf(a - mnew);
      float pc = __expf(c - mnew);
      float ps = pa + pc;
#pragma unroll
      for (int o = 8; o > 0; o >>= 1) ps += __shfl_xor(ps, o, 32);
      srun[r] = srun[r] * alpha + ps;
      mrun[r] = mnew;
      s01[0][r] = pa;
      s01[1][r] = pc;
#pragma unroll
      for (int t = 0; t < 4; ++t) oacc[t][r] *= alpha;
    }

    int mrow_off = (lane < 16) ? 0 : 8;
#pragma unroll
    for (int r = 0; r < 8; ++r) {
      pbuf[(r + mrow_off) * 32 + nk]      = (h16)s01[0][r];
      pbuf[(r + mrow_off) * 32 + 16 + nk] = (h16)s01[1][r];
    }
    __builtin_amdgcn_wave_barrier();
    asm volatile("s_wait_dscnt 0x0" ::: "memory");

    v16h ap;
    {
      const u32* pb = (const u32*)&pbuf[(lane & 15) * 32];
      u32* pu = (u32*)&ap;
#pragma unroll
      for (int v = 0; v < 4; ++v) {
        pu[v]     = pb[kb + v];
        pu[v + 4] = pb[kb + 4 + v];
      }
    }
    asm volatile("" ::: "memory");

#pragma unroll
    for (int nt = 0; nt < 4; ++nt) {
      v16h vf;
      h16* vh = (h16*)&vf;
#pragma unroll
      for (int v = 0; v < 8; ++v) {
        int kk = khalf + 2 * v;
        vh[2 * v]     = base_bh[(long long)(kt0 + kk)     * (3 * DMODEL) + 2 * DMODEL + h * HDIM + nt * 16 + nk];
        vh[2 * v + 1] = base_bh[(long long)(kt0 + kk + 1) * (3 * DMODEL) + 2 * DMODEL + h * HDIM + nt * 16 + nk];
      }
      oacc[nt] = __builtin_amdgcn_wmma_f32_16x16x32_f16(false, ap, false, vf, (short)0, oacc[nt], false, false);
    }
  }

  int mrow_off = (lane < 16) ? 0 : 8;
#pragma unroll
  for (int r = 0; r < 8; ++r) {
    float inv = 1.0f / srun[r];
    long long row = rowbase + mrow_off + r;
#pragma unroll
    for (int nt = 0; nt < 4; ++nt)
      O[row * DMODEL + h * HDIM + nt * 16 + nk] = (h16)(oacc[nt][r] * inv);
  }
}

/* ----------------------- classifier + deconv head ------------------------- */

__global__ __launch_bounds__(256)
void head_kernel(const float* __restrict__ X, const float* __restrict__ cls_w,
                 const float* __restrict__ cls_b, const float* __restrict__ dc_w,
                 const float* __restrict__ dc_b, float* __restrict__ out) {
  int b = blockIdx.y;
  int n = blockIdx.x;
  int tid = threadIdx.x;
  int wave = tid >> 5, lane = tid & 31;
  __shared__ float red[2][8];
  __shared__ float logit[2];
  const float* xrow = X + ((long long)b * NPAD + 1 + n) * DMODEL;
  float p0 = 0.0f, p1 = 0.0f;
  for (int d = tid; d < DMODEL; d += 256) {
    float xv = xrow[d];
    p0 += xv * cls_w[d * NCLS + 0];
    p1 += xv * cls_w[d * NCLS + 1];
  }
#pragma unroll
  for (int o = 16; o > 0; o >>= 1) { p0 += __shfl_xor(p0, o, 32); p1 += __shfl_xor(p1, o, 32); }
  if (lane == 0) { red[0][wave] = p0; red[1][wave] = p1; }
  __syncthreads();
  if (tid == 0) {
    float a = 0.0f, c = 0.0f;
#pragma unroll
    for (int w = 0; w < 8; ++w) { a += red[0][w]; c += red[1][w]; }
    logit[0] = a + cls_b[0];
    logit[1] = c + cls_b[1];
  }
  __syncthreads();
  int hh = tid >> 4, ww = tid & 15;
  float z0 = logit[0], z1 = logit[1];
#pragma unroll
  for (int o = 0; o < NCLS; ++o) {
    float v = z0 * dc_w[((0 * NCLS + o) * PHH + hh) * PWW + ww] +
              z1 * dc_w[((1 * NCLS + o) * PHH + hh) * PWW + ww] + dc_b[o];
    out[(((long long)b * NCLS + o) * (NPATCH * PHH) + (long long)n * PHH + hh) * PWW + ww] = v;
  }
}

/* ------------------------------- launcher ---------------------------------- */

extern "C" void kernel_launch(void* const* d_in, const int* in_sizes, int n_in,
                              void* d_out, int out_size, void* d_ws, size_t ws_size,
                              hipStream_t stream) {
  (void)in_sizes; (void)n_in; (void)out_size; (void)ws_size;
  const float* images  = (const float*)d_in[0];
  const float* patch_w = (const float*)d_in[2];
  const float* patch_b = (const float*)d_in[3];
  const float* cls_tok = (const float*)d_in[4];
  const float* ln1_g = (const float*)d_in[5];
  const float* ln1_b = (const float*)d_in[6];
  const float* wq = (const float*)d_in[7];
  const float* bq = (const float*)d_in[8];
  const float* wk = (const float*)d_in[9];
  const float* bk = (const float*)d_in[10];
  const float* wv = (const float*)d_in[11];
  const float* bvp = (const float*)d_in[12];
  const float* wo = (const float*)d_in[13];
  const float* bo = (const float*)d_in[14];
  const float* ln2_g = (const float*)d_in[15];
  const float* ln2_b = (const float*)d_in[16];
  const float* w1 = (const float*)d_in[17];
  const float* b1 = (const float*)d_in[18];
  const float* w2 = (const float*)d_in[19];
  const float* b2 = (const float*)d_in[20];
  const float* cls_w = (const float*)d_in[21];
  const float* cls_b = (const float*)d_in[22];
  const float* dc_w = (const float*)d_in[23];
  const float* dc_b = (const float*)d_in[24];
  float* out = (float*)d_out;

  char* w = (char*)d_ws;
  size_t off = 0;
  auto alloc = [&](size_t bytes) { void* p = w + off; off = (off + bytes + 255) & ~(size_t)255; return p; };
  float* X      = (float*)alloc((size_t)MROWS * DMODEL * 4);
  h16*   H16    = (h16*)  alloc((size_t)MROWS * DMODEL * 2);
  h16*   QKV16  = (h16*)  alloc((size_t)MROWS * 3 * DMODEL * 2);
  h16*   O16    = (h16*)  alloc((size_t)MROWS * DMODEL * 2);
  h16*   M116   = (h16*)  alloc((size_t)MROWS * DFF * 2);
  h16*   Wqkv16 = (h16*)  alloc((size_t)NLAYER * DMODEL * 3 * DMODEL * 2);
  float* Bqkv   = (float*)alloc((size_t)NLAYER * 3 * DMODEL * 4);
  h16*   Wo16   = (h16*)  alloc((size_t)NLAYER * DMODEL * DMODEL * 2);
  h16*   W116   = (h16*)  alloc((size_t)NLAYER * DMODEL * DFF * 2);
  h16*   W216   = (h16*)  alloc((size_t)NLAYER * DFF * DMODEL * 2);

  {
    long long t = (long long)NLAYER * DMODEL * 3 * DMODEL;
    pack_qkv_kernel<<<(unsigned)((t + 255) / 256), 256, 0, stream>>>(wq, wk, wv, bq, bk, bvp, Wqkv16, Bqkv);
    int n1 = NLAYER * DMODEL * DMODEL;
    cvt_f16_kernel<<<(n1 + 255) / 256, 256, 0, stream>>>(wo, Wo16, n1);
    int n2 = NLAYER * DMODEL * DFF;
    cvt_f16_kernel<<<(n2 + 255) / 256, 256, 0, stream>>>(w1, W116, n2);
    cvt_f16_kernel<<<(n2 + 255) / 256, 256, 0, stream>>>(w2, W216, n2);
  }

  embed_kernel<<<MROWS, DMODEL, 0, stream>>>(images, patch_w, patch_b, cls_tok, X);

  dim3 gQKV(3 * DMODEL / BN, MROWS / BM);
  dim3 gD(DMODEL / BN, MROWS / BM);
  dim3 gFF(DFF / BN, MROWS / BM);

  for (int l = 0; l < NLAYER; ++l) {
    ln_kernel<<<MROWS, 128, 0, stream>>>(X, ln1_g + l * DMODEL, ln1_b + l * DMODEL, H16);
    gemm_kernel<<<gQKV, 256, 0, stream>>>(H16, Wqkv16 + (size_t)l * DMODEL * 3 * DMODEL,
                                          Bqkv + l * 3 * DMODEL, nullptr, nullptr, QKV16,
                                          3 * DMODEL, DMODEL, EP_F16);
    attn_kernel<<<BATCH * NHEAD * 17, 128, 0, stream>>>(QKV16, O16);
    gemm_kernel<<<gD, 256, 0, stream>>>(O16, Wo16 + (size_t)l * DMODEL * DMODEL,
                                        bo + l * DMODEL, X, X, nullptr,
                                        DMODEL, DMODEL, EP_F32_RESID);
    ln_kernel<<<MROWS, 128, 0, stream>>>(X, ln2_g + l * DMODEL, ln2_b + l * DMODEL, H16);
    gemm_kernel<<<gFF, 256, 0, stream>>>(H16, W116 + (size_t)l * DMODEL * DFF,
                                         b1 + l * DFF, nullptr, nullptr, M116,
                                         DFF, DMODEL, EP_F16_GELU);
    gemm_kernel<<<gD, 256, 0, stream>>>(M116, W216 + (size_t)l * DFF * DMODEL,
                                        b2 + l * DMODEL, X, X, nullptr,
                                        DMODEL, DFF, EP_F32_RESID);
  }

  dim3 gHead(NPATCH, BATCH);
  head_kernel<<<gHead, 256, 0, stream>>>(X, cls_w, cls_b, dc_w, dc_b, out);
}
